// OldTensorProductConvLayer_44220983280193
// MI455X (gfx1250) — matrix operand order repaired and checked by reading, not compiled, not run on hardware
//
#include <hip/hip_runtime.h>
#include <stdint.h>

// ---------------- types ----------------
typedef __attribute__((ext_vector_type(16))) __bf16 v16bf;
typedef __attribute__((ext_vector_type(8)))  float  v8f;

#define NS 32
#define NV 8
#define HID 64
#define NEF 64
#define WNUM 1600          // 1024 + 256 + 64 + 256
#define TILE_E 64          // edges per block
#define NCH 64             // w-columns per chunk
#define NCHP 68            // padded row stride for sW (bank-conflict avoidance)
#define NCHUNKS (WNUM / NCH)   // 25
#define INV40f  0.15811388300841897f
#define INV120f 0.09128709291752768f
#define EPSf    1e-5f

static __device__ __forceinline__ unsigned short f2bf(float x) {
    unsigned int u = __builtin_bit_cast(unsigned int, x);
    unsigned int r = u + 0x7fffu + ((u >> 16) & 1u);
    return (unsigned short)(r >> 16);
}

union V16U { v16bf v; unsigned short s[16]; uint4 q[2]; };

// A-matrix (16x32 bf16) loader from a row-major bf16 [64][64] LDS buffer.
// lane 0-15: M = mt*16+lane, K = kstep*32 + {0..7, 16..23}
// lane16-31: M = mt*16+lane-16, K = kstep*32 + {8..15, 24..31}
static __device__ __forceinline__ v16bf load_a_tile(const unsigned short* S,
                                                    int mt, int kstep, int lane) {
    int m    = mt * 16 + (lane & 15);
    int half = lane >> 4;
    const unsigned short* row = S + m * 64 + kstep * 32 + half * 8;
    V16U u;
#pragma unroll
    for (int e = 0; e < 8; ++e) u.s[e] = row[e];
#pragma unroll
    for (int e = 0; e < 8; ++e) u.s[8 + e] = row[16 + e];
    return u.v;
}

// B-matrix (32x16 bf16) loader from pre-packed global tiles:
// tile layout = 32 lanes x 16 contiguous bf16 (lane = f(N,Khalf), elems = K)
static __device__ __forceinline__ v16bf load_b_tile(const unsigned short* B,
                                                    int tile, int lane) {
    const uint4* p = (const uint4*)(B + (size_t)(tile * 32 + lane) * 16);
    V16U u;
    u.q[0] = p[0];
    u.q[1] = p[1];
    return u.v;
}

// ---------------- prep: repack fc weights into bf16 B-tile layout ----------
__global__ void prep_pack_b(const float* __restrict__ W, unsigned short* __restrict__ out,
                            int ncols, int ntilesN, int total) {
    int idx = blockIdx.x * blockDim.x + threadIdx.x;
    if (idx >= total) return;
    int tile = idx >> 9;          // 512 elems per tile
    int r    = idx & 511;
    int l    = r >> 4;            // lane
    int e    = r & 15;            // element
    int kstep = tile / ntilesN;
    int nt    = tile % ntilesN;
    int k = kstep * 32 + 8 * (l >> 4) + (e & 7) + ((e >> 3) << 4);
    int n = nt * 16 + (l & 15);
    out[idx] = f2bf(W[k * ncols + n]);
}

__global__ void zero_f32(float* p, int n) {
    int i = blockIdx.x * blockDim.x + threadIdx.x;
    for (; i < n; i += gridDim.x * blockDim.x) p[i] = 0.f;
}

// ---------------- main fused edge kernel ----------------
__global__ void __launch_bounds__(256)
edge_kernel(const float* __restrict__ edge_attr,
            const long long* __restrict__ edge_index,
            const float* __restrict__ node_attr,
            const float* __restrict__ edge_sh,
            const unsigned short* __restrict__ w1bf,
            const unsigned short* __restrict__ w2bf,
            const float* __restrict__ fc1_b,
            const float* __restrict__ fc2_b,
            float* __restrict__ gsums, float* __restrict__ gcnt,
            int E, int N) {
    // LDS layout (40.5 KB):
    //  [0,17408)     : sA (64x64 u16, 8KB)  -- aliased by sW (64x68 f32) in stage 3
    //  [17408,25600) : sH (64x64 u16)
    //  [25600,39936) : sX (64x56 f32)
    //  [39936,40960) : sSH (64x4 f32)
    //  [40960,41216) : sSRC (64 i32)
    //  [41216,41472) : sDST (64 i32)
    __shared__ __attribute__((aligned(16))) char smem[41472];
    unsigned short* sA  = (unsigned short*)smem;
    float*          sW  = (float*)smem;                    // stride NCHP
    unsigned short* sH  = (unsigned short*)(smem + 17408);
    float*          sX  = (float*)(smem + 25600);
    float*          sSH = (float*)(smem + 39936);
    int*            sSRC= (int*)(smem + 40960);
    int*            sDST= (int*)(smem + 41216);

    const int tid  = threadIdx.x;
    const int lane = tid & 31;
    const int wave = tid >> 5;
    const int e0   = blockIdx.x * TILE_E;

    // ---- phase 0a: stage inputs ----
    for (int t = tid; t < TILE_E * NEF; t += 256) {
        int e = t >> 6, f = t & 63;
        int ge = e0 + e;
        sA[t] = (ge < E) ? f2bf(edge_attr[(size_t)ge * NEF + f]) : (unsigned short)0;
    }
    for (int t = tid; t < TILE_E * 4; t += 256) {
        int e = t >> 2, c = t & 3;
        int ge = e0 + e;
        sSH[t] = (ge < E) ? edge_sh[(size_t)ge * 4 + c] : 0.f;
    }
    for (int t = tid; t < TILE_E; t += 256) {
        int ge = e0 + t;
        sSRC[t] = (ge < E) ? (int)edge_index[ge] : 0;
        sDST[t] = (ge < E) ? (int)edge_index[(size_t)E + ge] : 0;
    }
    __syncthreads();
    for (int t = tid; t < TILE_E * 56; t += 256) {
        int e = t / 56, f = t % 56;
        int ge = e0 + e;
        sX[t] = (ge < E) ? node_attr[(size_t)sDST[e] * 56 + f] : 0.f;
    }
    __syncthreads();

    // ---- stage 2: h = relu(A @ W1 + b1) -> sH (bf16) ----
    // 16 tiles (4 mt x 4 nt), 2 per wave; executed convergently (EXEC all 1s)
#pragma unroll
    for (int jj = 0; jj < 2; ++jj) {
        int job = wave * 2 + jj;
        int mt = job >> 2, nt = job & 3;
        v8f c = {};
        v16bf a0 = load_a_tile(sA, mt, 0, lane);
        v16bf b0 = load_b_tile(w1bf, 0 * 4 + nt, lane);
        c = __builtin_amdgcn_wmma_f32_16x16x32_bf16(false, a0, false, b0, (short)0, c, false, false);
        v16bf a1 = load_a_tile(sA, mt, 1, lane);
        v16bf b1 = load_b_tile(w1bf, 1 * 4 + nt, lane);
        c = __builtin_amdgcn_wmma_f32_16x16x32_bf16(false, a1, false, b1, (short)0, c, false, false);
        int n    = nt * 16 + (lane & 15);
        int half = lane >> 4;
        float bias = fc1_b[n];
#pragma unroll
        for (int v = 0; v < 8; ++v) {
            float h = fmaxf(c[v] + bias, 0.f);
            int m = mt * 16 + v + 8 * half;
            sH[m * 64 + n] = f2bf(h);
        }
    }
    __syncthreads();   // sH ready; sA region now dead -> reused as sW

    // ---- stage 3: stream w = H @ W2 + b2 in 64-col chunks, gather-fold TP ----
    // Ownership: thread -> edge e = tid>>2, channels [c0, c0+14) with c0 = (tid&3)*14.
    // Exclusive ownership -> register accumulators, zero atomics in the fold.
    const int e    = tid >> 2;
    const int c0   = (tid & 3) * 14;
    const float* xe = sX + e * 56;
    const float s0v  = sSH[e * 4 + 0];
    const float s1v0 = sSH[e * 4 + 1];
    const float s1v1 = sSH[e * 4 + 2];
    const float s1v2 = sSH[e * 4 + 3];
    float acc[14];
#pragma unroll
    for (int k = 0; k < 14; ++k) acc[k] = 0.f;
    int cc3[14], i3[14];
#pragma unroll
    for (int k = 0; k < 14; ++k) {
        int d = c0 + k - 32;
        cc3[k] = (d >= 0) ? d / 3 : 0;
        i3[k]  = (d >= 0) ? d % 3 : 0;
    }

    for (int chunk = 0; chunk < NCHUNKS; ++chunk) {
        if (chunk + 1 < NCHUNKS)   // pull next chunk's B tiles toward L2/WGP$
            __builtin_prefetch(w2bf + (size_t)((chunk + 1) * 4) * 512, 0, 3);
        // -- WMMA: 16 tiles (4 mt x 4 nt), 2 per wave, convergent --
        // fc2 bias is folded into the epilogue store (lane knows its column n),
        // so the gather below reads bias-inclusive w values directly.
#pragma unroll
        for (int jj = 0; jj < 2; ++jj) {
            int job = wave * 2 + jj;
            int mt = job >> 2, nt = job & 3;
            v8f c = {};
            v16bf a0 = load_a_tile(sH, mt, 0, lane);
            v16bf b0 = load_b_tile(w2bf, 0 * 100 + chunk * 4 + nt, lane);
            c = __builtin_amdgcn_wmma_f32_16x16x32_bf16(false, a0, false, b0, (short)0, c, false, false);
            v16bf a1 = load_a_tile(sH, mt, 1, lane);
            v16bf b1 = load_b_tile(w2bf, 1 * 100 + chunk * 4 + nt, lane);
            c = __builtin_amdgcn_wmma_f32_16x16x32_bf16(false, a1, false, b1, (short)0, c, false, false);
            int n    = nt * 16 + (lane & 15);
            int half = lane >> 4;
            float bias = fc2_b[chunk * NCH + n];
#pragma unroll
            for (int v = 0; v < 8; ++v)
                sW[(mt * 16 + v + 8 * half) * NCHP + n] = c[v] + bias;
        }
        __syncthreads();

        const float* we = sW + e * NCHP;
        if (chunk < 16) {
            // w00 region: j = chunk*64 + jl ; u = 2*chunk + (jl>>5), col = jl&31
            // out0[c] += s0/sqrt40 * x0[u] * w
            float a0 = xe[2 * chunk]     * s0v * INV40f;
            float a1 = xe[2 * chunk + 1] * s0v * INV40f;
#pragma unroll
            for (int k = 0; k < 14; ++k) {
                int c = c0 + k;
                if (c < 32)
                    acc[k] += a0 * we[c] + a1 * we[32 + c];
            }
        } else if (chunk < 20) {
            // w01 region: u = (chunk-16)*8 + (jl>>3), col = jl&7
            // out1[cc,i] += s1[i]/sqrt40 * sum_u x0[u]*w
            int u0 = (chunk - 16) * 8;
#pragma unroll
            for (int k = 0; k < 14; ++k) {
                int c = c0 + k;
                if (c >= 32) {
                    int cc = cc3[k];
                    float s1i = (i3[k] == 0) ? s1v0 : ((i3[k] == 1) ? s1v1 : s1v2);
                    float sum = 0.f;
#pragma unroll
                    for (int du = 0; du < 8; ++du)
                        sum += xe[u0 + du] * we[du * 8 + cc];
                    acc[k] += s1i * INV40f * sum;
                }
            }
        } else if (chunk == 20) {
            // w10 region (exactly one chunk): u = jl>>3, col = jl&7
            // out1[cc,i] += s0/sqrt40 * sum_u x1[u,i]*w
#pragma unroll
            for (int k = 0; k < 14; ++k) {
                int c = c0 + k;
                if (c >= 32) {
                    int cc = cc3[k], i = i3[k];
                    float sum = 0.f;
#pragma unroll
                    for (int du = 0; du < 8; ++du)
                        sum += xe[32 + du * 3 + i] * we[du * 8 + cc];
                    acc[k] += s0v * INV40f * sum;
                }
            }
        } else {
            // w11 region: u = 2*(chunk-21) + (jl>>5), col = jl&31
            // out0[c] += (x1[u]·s1)/sqrt120 * w
            int u0 = 2 * (chunk - 21);
            float t0 = (xe[32 + u0 * 3] * s1v0 + xe[33 + u0 * 3] * s1v1 +
                        xe[34 + u0 * 3] * s1v2) * INV120f;
            float t1 = (xe[35 + u0 * 3] * s1v0 + xe[36 + u0 * 3] * s1v1 +
                        xe[37 + u0 * 3] * s1v2) * INV120f;
#pragma unroll
            for (int k = 0; k < 14; ++k) {
                int c = c0 + k;
                if (c < 32)
                    acc[k] += t0 * we[c] + t1 * we[32 + c];
            }
        }
        __syncthreads();
    }

    // ---- scatter: segment-sum over edge_src (global atomics only) ----
    if (e0 + e < E) {
        int src = sSRC[e];
#pragma unroll
        for (int k = 0; k < 14; ++k)
            atomicAdd(&gsums[(size_t)src * 56 + c0 + k], acc[k]);
        if ((tid & 3) == 0) atomicAdd(&gcnt[src], 1.0f);
    }
}

// ---------------- node stage A: mean + residual + channel stats -----------
__global__ void __launch_bounds__(256)
node_stage_a(const float* __restrict__ gsums, const float* __restrict__ gcnt,
             const float* __restrict__ node_attr, float* __restrict__ nodeout,
             float* __restrict__ gstats, int N) {
    __shared__ float red[72];   // [0,32)=sum sc, [32,64)=sum sc^2, [64,72)=sum vnorm
    int tid = threadIdx.x;
    if (tid < 72) red[tid] = 0.f;
    __syncthreads();
    int n = blockIdx.x * blockDim.x + tid;
    if (n < N) {
        float inv = 1.0f / fmaxf(gcnt[n], 1.0f);
        float o[56];
#pragma unroll
        for (int c = 0; c < 56; ++c) {
            o[c] = gsums[(size_t)n * 56 + c] * inv + node_attr[(size_t)n * 56 + c];
            nodeout[(size_t)n * 56 + c] = o[c];
        }
#pragma unroll
        for (int c = 0; c < 32; ++c) {
            atomicAdd(&red[c], o[c]);
            atomicAdd(&red[32 + c], o[c] * o[c]);
        }
#pragma unroll
        for (int v = 0; v < 8; ++v) {
            float a = o[32 + v * 3], b = o[33 + v * 3], cc = o[34 + v * 3];
            atomicAdd(&red[64 + v], (a * a + b * b + cc * cc) * (1.0f / 3.0f));
        }
    }
    __syncthreads();
    if (tid < 72) atomicAdd(&gstats[tid], red[tid]);
}

// ---------------- node stage B: batch-norm finalize ----------------
__global__ void node_stage_b(const float* __restrict__ nodeout,
                             const float* __restrict__ gstats,
                             const float* __restrict__ g0, const float* __restrict__ b0,
                             const float* __restrict__ g1,
                             float* __restrict__ out, int N) {
    int idx = blockIdx.x * blockDim.x + threadIdx.x;
    if (idx >= N * 56) return;
    int c = idx % 56;
    float invN = 1.0f / (float)N;
    float x = nodeout[idx];
    float r;
    if (c < 32) {
        float m   = gstats[c] * invN;
        float var = gstats[32 + c] * invN - m * m;
        r = (x - m) * rsqrtf(var + EPSf) * g0[c] + b0[c];
    } else {
        int v = (c - 32) / 3;
        float vn = gstats[64 + v] * invN;
        r = x * rsqrtf(vn + EPSf) * g1[v];
    }
    out[idx] = r;
}

// ---------------- host launcher ----------------
extern "C" void kernel_launch(void* const* d_in, const int* in_sizes, int n_in,
                              void* d_out, int out_size, void* d_ws, size_t ws_size,
                              hipStream_t stream) {
    const float*     node_attr = (const float*)d_in[0];
    const float*     edge_attr = (const float*)d_in[1];
    const float*     edge_sh   = (const float*)d_in[2];
    const long long* edge_index= (const long long*)d_in[3];
    const float*     fc1_w     = (const float*)d_in[4];
    const float*     fc1_b     = (const float*)d_in[5];
    const float*     fc2_w     = (const float*)d_in[6];
    const float*     fc2_b     = (const float*)d_in[7];
    const float*     bn_g0     = (const float*)d_in[8];
    const float*     bn_b0     = (const float*)d_in[9];
    const float*     bn_g1     = (const float*)d_in[10];

    const int N = in_sizes[0] / 56;
    const int E = in_sizes[1] / NEF;

    char* ws = (char*)d_ws;
    float*          gsums   = (float*)(ws + 0);                 // N*56 f32
    float*          gcnt    = (float*)(ws + (size_t)N * 56 * 4);// N f32
    float*          nodeout = (float*)(ws + (size_t)N * 57 * 4);// N*56 f32
    float*          gstats  = (float*)(ws + (size_t)N * 113 * 4);// 72 f32 (+pad)
    unsigned short* w1bf    = (unsigned short*)(ws + (size_t)N * 113 * 4 + 512);
    unsigned short* w2bf    = w1bf + 4096;

    // repack weights to bf16 WMMA-B tile layout
    prep_pack_b<<<(4096 + 255) / 256, 256, 0, stream>>>(fc1_w, w1bf, HID, 4, 4096);
    prep_pack_b<<<(102400 + 255) / 256, 256, 0, stream>>>(fc2_w, w2bf, WNUM, 100, 102400);

    // zero accumulators (must happen every launch)
    zero_f32<<<256, 256, 0, stream>>>(gsums, N * 56);
    zero_f32<<<64, 256, 0, stream>>>(gcnt, N);
    zero_f32<<<1, 256, 0, stream>>>(gstats, 72);

    // fused MLP + tensor-product + scatter
    int blocks = (E + TILE_E - 1) / TILE_E;
    edge_kernel<<<blocks, 256, 0, stream>>>(edge_attr, edge_index, node_attr, edge_sh,
                                            w1bf, w2bf, fc1_b, fc2_b, gsums, gcnt, E, N);

    // node-side finalize
    node_stage_a<<<(N + 255) / 256, 256, 0, stream>>>(gsums, gcnt, node_attr, nodeout,
                                                      gstats, N);
    node_stage_b<<<(N * 56 + 255) / 256, 256, 0, stream>>>(nodeout, gstats, bn_g0, bn_b0,
                                                           bn_g1, (float*)d_out, N);
}
